// CustomLinear_8229157339512
// MI455X (gfx1250) — matrix operand order repaired and checked by reading, not compiled, not run on hardware
//
#include <hip/hip_runtime.h>
#include <hip/hip_bf16.h>

// ---------------------------------------------------------------------------
// BFP8 (group=16, m_bit=8) quant-dequant into bf16, then bf16 WMMA GEMM + bias
// MI455X / gfx1250: wave32, v_wmma_f32_16x16x32_bf16, async LDS double-buffer
// ---------------------------------------------------------------------------

typedef __bf16  v16bf __attribute__((ext_vector_type(16)));
typedef float   v8f   __attribute__((ext_vector_type(8)));

#define BM 128
#define BN 128
#define BK 32
#define APAD 8   // (32+8)*2  = 80B rows (16B-aligned chunks)
#define BPAD 8   // (128+8)*2 = 272B rows

#if defined(__HIP_DEVICE_COMPILE__) && __has_builtin(__builtin_amdgcn_global_load_async_to_lds_b128)
#define HAVE_ASYNC_LDS 1
#else
#define HAVE_ASYNC_LDS 0
#endif

// 16-byte global -> LDS copy. Async (ASYNCcnt path) when available.
__device__ __forceinline__ void copy16(void* lds, const void* g) {
#if HAVE_ASYNC_LDS
    typedef int v4i_ __attribute__((vector_size(16)));
    __builtin_amdgcn_global_load_async_to_lds_b128(
        (__attribute__((address_space(1))) v4i_*)g,
        (__attribute__((address_space(3))) v4i_*)lds, 0, 0);
#else
    *(float4*)lds = *(const float4*)g;
#endif
}

// Wait until at most CNT of this wave's async copies remain in flight.
template <int CNT>
__device__ __forceinline__ void wait_async() {
#if HAVE_ASYNC_LDS
#if __has_builtin(__builtin_amdgcn_s_wait_asynccnt)
    __builtin_amdgcn_s_wait_asynccnt(CNT);
#else
    asm volatile("s_wait_asynccnt %0" ::"i"(CNT) : "memory");
#endif
#endif
}

// ---------------------------------------------------------------------------
// BFP quantize one group of 16 contiguous fp32 -> 16 bf16 (exact: |m|<=127)
// ---------------------------------------------------------------------------
__device__ __forceinline__ void quant_group16(const float* __restrict__ src,
                                              unsigned short* __restrict__ dst) {
    const float4* p = (const float4*)src;
    float4 c0 = p[0], c1 = p[1], c2 = p[2], c3 = p[3];
    float v[16] = {c0.x, c0.y, c0.z, c0.w, c1.x, c1.y, c1.z, c1.w,
                   c2.x, c2.y, c2.z, c2.w, c3.x, c3.y, c3.z, c3.w};
    float mx = 0.f;
#pragma unroll
    for (int i = 0; i < 16; ++i) mx = fmaxf(mx, fabsf(v[i]));
    mx = fmaxf(mx, 1e-38f);
    int e = (int)((__float_as_uint(mx) >> 23) & 0xFFu) - 127;  // floor(log2) for normals
    float scale = exp2f((float)(e - 7));
    float inv   = exp2f((float)(7 - e));
#pragma unroll
    for (int i = 0; i < 16; ++i) {
        float q = fminf(fmaxf(rintf(v[i] * inv), -127.f), 127.f) * scale;
        dst[i] = (unsigned short)(__float_as_uint(q) >> 16);  // exact bf16
    }
}

// qx: [M,K] fp32 -> [M,K] bf16 (groups contiguous along K)
__global__ __launch_bounds__(256) void bfp_quant_x(const float* __restrict__ x,
                                                   unsigned short* __restrict__ q,
                                                   long long nGroups) {
    long long g = (long long)blockIdx.x * blockDim.x + threadIdx.x;
    if (g >= nGroups) return;
    __align__(16) unsigned short tmp[16];
    quant_group16(x + g * 16, tmp);
    float4* o = (float4*)(q + g * 16);
    o[0] = *(float4*)&tmp[0];
    o[1] = *(float4*)&tmp[8];
}

// weight: [N,K] fp32 -> K-major bf16 [K,N] (so GEMM B tiles are contiguous)
__global__ __launch_bounds__(256) void bfp_quant_w_t(const float* __restrict__ w,
                                                     unsigned short* __restrict__ qT,
                                                     int N, int K, long long nGroups) {
    long long g = (long long)blockIdx.x * blockDim.x + threadIdx.x;
    if (g >= nGroups) return;
    int kgPerRow = K / 16;
    int kg = (int)(g % kgPerRow);
    int n  = (int)(g / kgPerRow);
    __align__(16) unsigned short tmp[16];
    quant_group16(w + (size_t)n * K + (size_t)kg * 16, tmp);
#pragma unroll
    for (int i = 0; i < 16; ++i)
        qT[(size_t)(kg * 16 + i) * N + n] = tmp[i];
}

// ---------------------------------------------------------------------------
// GEMM: out[m,n] = sum_k qx[m,k] * qwT[k,n] + bias[n]
// 128x128x32 tiles, 8 waves, each wave = 2x4 grid of 16x16 WMMA tiles.
// Double-buffered LDS; async copies for tile i+1 overlap WMMAs of tile i.
// ---------------------------------------------------------------------------
__global__ __launch_bounds__(256) void bfp_gemm(const unsigned short* __restrict__ qx,
                                                const unsigned short* __restrict__ qwT,
                                                const float* __restrict__ bias,
                                                float* __restrict__ out,
                                                int M, int N, int K) {
    __shared__ unsigned short As[2][BM][BK + APAD];  // A tiles, [m][k]
    __shared__ unsigned short Bs[2][BK][BN + BPAD];  // B tiles, [k][n]

    const int t     = threadIdx.x;
    const int mBase = blockIdx.y * BM;
    const int nBase = blockIdx.x * BN;
    const int wave  = t >> 5;            // wave32: 8 waves per block
    const int lane  = t & 31;
    const int wm    = (wave & 3) * 32;   // 4 waves along M
    const int wn    = (wave >> 2) * 64;  // 2 waves along N

    // Staging assignment: every thread moves exactly 4x16B (no divergence).
    const int ar = t >> 2, ac = (t & 3) * 8;    // A rows ar, ar+64; 4 chunks/row
    const int br = t >> 4, bc = (t & 15) * 8;   // B rows br, br+16; 16 chunks/row

    union CAcc { v8f v; float f[8]; };
    CAcc acc[2][4];
#pragma unroll
    for (int i = 0; i < 2; ++i)
#pragma unroll
        for (int j = 0; j < 4; ++j)
#pragma unroll
            for (int r = 0; r < 8; ++r) acc[i][j].f[r] = 0.f;

    const int lr = lane & 15;   // row within 16x16 tile
    const int lh = lane >> 4;   // lane half selects K sub-range for A frag

    auto stage = [&](int kb, int buf) {
        copy16(&As[buf][ar][ac],      qx  + (size_t)(mBase + ar) * K + kb + ac);
        copy16(&As[buf][ar + 64][ac], qx  + (size_t)(mBase + ar + 64) * K + kb + ac);
        copy16(&Bs[buf][br][bc],      qwT + (size_t)(kb + br) * N + nBase + bc);
        copy16(&Bs[buf][br + 16][bc], qwT + (size_t)(kb + br + 16) * N + nBase + bc);
    };

    const int nTiles = K / BK;
    stage(0, 0);  // prologue: 4 async copies in flight

    for (int ti = 0; ti < nTiles; ++ti) {
        const int p = ti & 1;
        if (ti + 1 < nTiles) {
            stage((ti + 1) * BK, p ^ 1);  // prefetch next tile into other buffer
            wait_async<4>();              // tile ti's 4 copies done; next 4 in flight
        } else {
            wait_async<0>();
        }
        __syncthreads();                  // tile ti visible to all waves

        // ---- fragments per ISA VGPR layouts ----
        union Frag { v16bf v; float4 f4[2]; };
        Frag af[2], bfr[4];
#pragma unroll
        for (int i = 0; i < 2; ++i) {
            // A 16x32: lane<16 holds row m, K 0..7 & 16..23; lane>=16 K 8..15 & 24..31
            const unsigned short* ap = &As[p][wm + i * 16 + lr][lh * 8];
            af[i].f4[0] = *(const float4*)ap;
            af[i].f4[1] = *(const float4*)(ap + 16);
        }
#pragma unroll
        for (int j = 0; j < 4; ++j) {
            // B 32x16: lane holds column N = lane&15; K 0..7 & 16..23 (lane<16), 8..15 & 24..31 (lane>=16)
            const unsigned short* bp = &Bs[p][lh * 8][wn + j * 16 + lr];
#pragma unroll
            for (int s = 0; s < 16; ++s)
                bfr[j].v[s] = __builtin_bit_cast(__bf16, bp[(size_t)((s & 7) + (s >> 3) * 16) * (BN + BPAD)]);
        }

#pragma unroll
        for (int i = 0; i < 2; ++i)
#pragma unroll
            for (int j = 0; j < 4; ++j)
                acc[i][j].v = __builtin_amdgcn_wmma_f32_16x16x32_bf16(
                    false, af[i].v, false, bfr[j].v,
                    (short)0, acc[i][j].v, false, false);

        __syncthreads();  // all waves done reading buf p before it is restaged
    }

    // ---- epilogue: C layout lane<16 -> M=v, lane>=16 -> M=8+v; N = lane&15 ----
    const int col = lane & 15;
    const int rb  = (lane >> 4) * 8;
#pragma unroll
    for (int i = 0; i < 2; ++i)
#pragma unroll
        for (int j = 0; j < 4; ++j) {
            const int gn = nBase + wn + j * 16 + col;
            const float bv = bias[gn];
            float* op = out + (size_t)(mBase + wm + i * 16 + rb) * N + gn;
#pragma unroll
            for (int r = 0; r < 8; ++r)
                op[(size_t)r * N] = acc[i][j].f[r] + bv;
        }
}

// ---------------------------------------------------------------------------
extern "C" void kernel_launch(void* const* d_in, const int* in_sizes, int n_in,
                              void* d_out, int out_size, void* d_ws, size_t ws_size,
                              hipStream_t stream) {
    const float* x    = (const float*)d_in[0];  // [B,S,K] fp32
    const float* w    = (const float*)d_in[1];  // [N,K]   fp32
    const float* bias = (const float*)d_in[2];  // [N]     fp32
    float* out        = (float*)d_out;          // [B,S,N] fp32

    const int N = in_sizes[2];
    const int K = in_sizes[1] / N;
    const int M = in_sizes[0] / K;  // B*S

    unsigned short* qx  = (unsigned short*)d_ws;  // [M,K] bf16
    unsigned short* qwT = qx + (size_t)M * K;     // [K,N] bf16

    long long gx = (long long)M * K / 16;
    bfp_quant_x<<<dim3((unsigned)((gx + 255) / 256)), 256, 0, stream>>>(x, qx, gx);

    long long gw = (long long)N * K / 16;
    bfp_quant_w_t<<<dim3((unsigned)((gw + 255) / 256)), 256, 0, stream>>>(w, qwT, N, K, gw);

    bfp_gemm<<<dim3(N / BN, M / BM), 256, 0, stream>>>(qx, qwT, bias, out, M, N, K);
}
